// KVCacheMemory_62380105007502
// MI455X (gfx1250) — matrix operand order repaired and checked by reading, not compiled, not run on hardware
//
#include <hip/hip_runtime.h>
#include <hip/hip_bf16.h>

// ---------------------------------------------------------------------------
// KVCacheMemory dual-attention for gfx1250 (MI455X), wave32 + WMMA f16->f32.
// Attention K/V tiles are staged into LDS once per workgroup with CDNA5
// async-to-LDS loads (ASYNCcnt), double-buffered, bank-conflict padded.
// ---------------------------------------------------------------------------

#define NB 8
#define SS 4096
#define DD 192
#define NROWS (NB * SS)          // 32768
#define CH_READ 156
#define CH_WRITE 157
#define CH_READY 158

#define KPAD 200                 // padded K-tile row (halfs): 400B -> conflict-free
#define VPAD 40                  // padded V-tile row (halfs):  80B -> conflict-free

typedef __attribute__((ext_vector_type(16))) _Float16 v16h;
typedef __attribute__((ext_vector_type(8)))  _Float16 v8h;
typedef __attribute__((ext_vector_type(4)))  _Float16 v4h;
typedef __attribute__((ext_vector_type(8)))  float    v8f;
typedef int v4i __attribute__((vector_size(16)));

#define AS1 __attribute__((address_space(1)))
#define AS3 __attribute__((address_space(3)))

union V16 { v16h v; v8h h[2]; };

__device__ inline v8h ld8(const _Float16* p) { return *(const v8h*)p; }

// WMMA wrapper: D = A(16x32 f16) * B(32x16 f16) + C(16x16 f32)
__device__ inline v8f wmma_f16(v16h a, v16h b, v8f c) {
  return __builtin_amdgcn_wmma_f32_16x16x32_f16(
      /*neg_a=*/false, a, /*neg_b=*/false, b,
      /*c_mod=*/(short)0, c, /*reuse_a=*/false, /*reuse_b=*/false);
}

// A fragment 16x32 (MxK) from row-major src, tile (row0,k0).
__device__ inline v16h loadA(const _Float16* __restrict__ src, int ld, int row0,
                             int k0, int lane) {
  int m = lane & 15, hi = lane >> 4;
  const _Float16* r = src + (size_t)(row0 + m) * ld + k0 + hi * 8;
  V16 a; a.h[0] = ld8(r); a.h[1] = ld8(r + 16);
  return a.v;
}

// B fragment 32x16 (KxN) for C = A * W^T: B[k,n] = W[n0+n, k0+k], W row-major.
__device__ inline v16h loadB_NT(const _Float16* __restrict__ W, int ld, int n0,
                                int k0, int lane) {
  int n = lane & 15, hi = lane >> 4;
  const _Float16* r = W + (size_t)(n0 + n) * ld + k0 + hi * 8;
  V16 b; b.h[0] = ld8(r); b.h[1] = ld8(r + 16);
  return b.v;
}

// ---- CDNA5 async global->LDS copy (16 bytes per lane), ASYNCcnt tracked ----
__device__ inline void async_cp16(const _Float16* g, _Float16* l) {
#if __has_builtin(__builtin_amdgcn_global_load_async_to_lds_b128)
  __builtin_amdgcn_global_load_async_to_lds_b128((AS1 v4i*)g, (AS3 v4i*)l, 0, 0);
#else
  unsigned lo = (unsigned)(size_t)(AS3 void*)l;
  asm volatile("global_load_async_to_lds_b128 %0, %1, off"
               :: "v"(lo), "v"(g) : "memory");
#endif
}

#if __has_builtin(__builtin_amdgcn_s_wait_asynccnt)
#define ASYNC_WAIT(n) __builtin_amdgcn_s_wait_asynccnt(n)
#else
#define ASYNC_WAIT(n) asm volatile("s_wait_asynccnt %0" :: "i"(n) : "memory")
#endif

// ---------------------------------------------------------------------------
// fp32 -> f16 conversion (vectorized by 4)
// ---------------------------------------------------------------------------
__global__ __launch_bounds__(256) void cvt_f32_f16(const float* __restrict__ in,
                                                   _Float16* __restrict__ out,
                                                   int n) {
  int i = (blockIdx.x * 256 + threadIdx.x) * 4;
  if (i + 3 < n) {
    float4 f = *(const float4*)(in + i);
    v4h h; h.x = (_Float16)f.x; h.y = (_Float16)f.y;
    h.z = (_Float16)f.z; h.w = (_Float16)f.w;
    *(v4h*)(out + i) = h;
  } else {
    for (; i < n; ++i) out[i] = (_Float16)in[i];
  }
}

// ---------------------------------------------------------------------------
// Extract per-batch gate flags: fl[b]=read, fl[8+b]=write, fl[16+b]=ready
// ---------------------------------------------------------------------------
__global__ void flags_kernel(const float* __restrict__ x, float* __restrict__ fl) {
  int b = threadIdx.x;
  if (b < NB) {
    float r = x[(size_t)b * SS * DD + CH_READ];
    float w = x[(size_t)b * SS * DD + CH_WRITE];
    fl[b] = r; fl[NB + b] = w; fl[2 * NB + b] = r + w;
  }
}

// ---------------------------------------------------------------------------
// QKV projections (6 GEMMs selected by blockIdx.y).
// ---------------------------------------------------------------------------
__global__ __launch_bounds__(256)
void qkv_kernel(const _Float16* __restrict__ xh, const _Float16* __restrict__ wh,
                _Float16* __restrict__ q_r, _Float16* __restrict__ k_r,
                _Float16* __restrict__ vT_r, _Float16* __restrict__ q_w,
                _Float16* __restrict__ k_w, _Float16* __restrict__ vT_w) {
  int p = blockIdx.y;                      // 0:qr 1:kr 2:vr 3:qw 4:kw 5:vw
  int widx = p + (p >= 3 ? 1 : 0);         // skip Wo_r slot
  const _Float16* W = wh + (size_t)widx * DD * DD;
  _Float16* out = (p == 0) ? q_r : (p == 1) ? k_r : (p == 2) ? vT_r
                : (p == 3) ? q_w : (p == 4) ? k_w : vT_w;
  bool isV = (p == 2 || p == 5);
  float scale = (p == 0 || p == 3) ? 0.07216878364870323f : 1.0f;

  int lane = threadIdx.x & 31, wave = threadIdx.x >> 5;
  int row0 = (blockIdx.x * 8 + wave) * 16;
  int n = lane & 15, hi = lane >> 4;

  v16h a[6];
#pragma unroll
  for (int ks = 0; ks < 6; ++ks) a[ks] = loadA(xh, DD, row0, ks * 32, lane);

#pragma unroll
  for (int nt = 0; nt < 12; ++nt) {
    v8f acc = {};
#pragma unroll
    for (int ks = 0; ks < 6; ++ks)
      acc = wmma_f16(a[ks], loadB_NT(W, DD, nt * 16, ks * 32, lane), acc);
#pragma unroll
    for (int j = 0; j < 8; ++j) {
      int row = row0 + j + 8 * hi;
      int col = nt * 16 + n;
      _Float16 val = (_Float16)(acc[j] * scale);
      if (!isV) {
        out[(size_t)row * DD + col] = val;
      } else {
        int b = row >> 12, s = row & (SS - 1);
        out[((size_t)b * DD + col) * SS + s] = val;
      }
    }
  }
}

// ---------------------------------------------------------------------------
// Flash attention with async double-buffered K/V staging in LDS.
// Block = 8 waves sharing one K/V stream; each wave owns a 16-row Q tile.
// ---------------------------------------------------------------------------
__global__ __launch_bounds__(256)
void attn_kernel(const _Float16* __restrict__ q_r, const _Float16* __restrict__ k_r,
                 const _Float16* __restrict__ vT_r, _Float16* __restrict__ o_r,
                 const _Float16* __restrict__ q_w, const _Float16* __restrict__ k_w,
                 const _Float16* __restrict__ vT_w, _Float16* __restrict__ o_w) {
  int path = blockIdx.z;
  const _Float16* q  = path ? q_w  : q_r;
  const _Float16* k  = path ? k_w  : k_r;
  const _Float16* vT = path ? vT_w : vT_r;
  _Float16* oh       = path ? o_w  : o_r;

  int b = blockIdx.y;
  int lane = threadIdx.x & 31, wave = threadIdx.x >> 5;
  int tid = threadIdx.x;
  int qrow0 = (blockIdx.x * 8 + wave) * 16;     // row within batch
  const _Float16* qB = q  + (size_t)b * SS * DD;
  const _Float16* kB = k  + (size_t)b * SS * DD;
  const _Float16* vB = vT + (size_t)b * DD * SS;

  // padded LDS tiles (conflict-free B-fragment reads)
  __shared__ __align__(16) _Float16 kbuf[2][32 * KPAD];   // 25600 B
  __shared__ __align__(16) _Float16 vbuf[2][192 * VPAD];  // 30720 B
  __shared__ __align__(16) _Float16 pbuf[8][16 * 32];     //  8192 B
  _Float16* pw = pbuf[wave];

  int n = lane & 15, hi = lane >> 4;
  int mm = lane & 15;

  // Stage one 32-key chunk: K (32x192, contiguous) + V^T (192x32, strided rows).
  // 768 16-byte pieces each; 256 threads x 3 per tensor; 6 asyncs per wave.
  auto stage = [&](int kc, int buf) {
    int key0 = kc * 32;
#pragma unroll
    for (int i = 0; i < 3; ++i) {
      int c = i * 256 + tid;                  // 16B piece id for K
      int r = c / 24, w = c % 24;             // K row / piece-in-row
      async_cp16(kB + (size_t)key0 * DD + c * 8, &kbuf[buf][r * KPAD + w * 8]);
    }
#pragma unroll
    for (int i = 0; i < 3; ++i) {
      int c = i * 256 + tid;                  // 16B piece id for V
      int e = c >> 2, part = c & 3;
      async_cp16(vB + (size_t)e * SS + key0 + part * 8,
                 &vbuf[buf][e * VPAD + part * 8]);
    }
  };

  v16h qa[6];
#pragma unroll
  for (int ks = 0; ks < 6; ++ks) qa[ks] = loadA(qB, DD, qrow0, ks * 32, lane);

  v8f acc[12];
#pragma unroll
  for (int nt = 0; nt < 12; ++nt) acc[nt] = {};
  float mrow[8], lrow[8];
#pragma unroll
  for (int j = 0; j < 8; ++j) { mrow[j] = -1e30f; lrow[j] = 0.0f; }

  const int NCH = SS / 32;
  stage(0, 0);

  for (int kc = 0; kc < NCH; ++kc) {
    int cur = kc & 1;
    if (kc + 1 < NCH) {
      stage(kc + 1, cur ^ 1);
      ASYNC_WAIT(6);            // current chunk complete; next may be in flight
    } else {
      ASYNC_WAIT(0);
    }
    __syncthreads();

    const _Float16* kl = kbuf[cur];
    const _Float16* vl = vbuf[cur];

    // scores S = Q K^T (two 16-col tiles), B fragments from LDS
    v8f s0 = {}, s1 = {};
#pragma unroll
    for (int ks = 0; ks < 6; ++ks) {
      {
        const _Float16* r = kl + n * KPAD + ks * 32 + hi * 8;
        V16 kb; kb.h[0] = *(const v8h*)r; kb.h[1] = *(const v8h*)(r + 16);
        s0 = wmma_f16(qa[ks], kb.v, s0);
      }
      {
        const _Float16* r = kl + (16 + n) * KPAD + ks * 32 + hi * 8;
        V16 kb; kb.h[0] = *(const v8h*)r; kb.h[1] = *(const v8h*)(r + 16);
        s1 = wmma_f16(qa[ks], kb.v, s1);
      }
    }
    // online softmax per fragment row j (row = j + 8*hi, cols across lanes)
#pragma unroll
    for (int j = 0; j < 8; ++j) {
      float t = fmaxf(s0[j], s1[j]);
      t = fmaxf(t, __shfl_xor(t, 1, 32));
      t = fmaxf(t, __shfl_xor(t, 2, 32));
      t = fmaxf(t, __shfl_xor(t, 4, 32));
      t = fmaxf(t, __shfl_xor(t, 8, 32));
      float mn = fmaxf(mrow[j], t);
      float cf = __expf(mrow[j] - mn);
      float p0 = __expf(s0[j] - mn);
      float p1 = __expf(s1[j] - mn);
      float rs = p0 + p1;
      rs += __shfl_xor(rs, 1, 32);
      rs += __shfl_xor(rs, 2, 32);
      rs += __shfl_xor(rs, 4, 32);
      rs += __shfl_xor(rs, 8, 32);
      lrow[j] = lrow[j] * cf + rs;
      mrow[j] = mn;
#pragma unroll
      for (int nt = 0; nt < 12; ++nt) acc[nt][j] = acc[nt][j] * cf;
      int r = j + 8 * hi;
      pw[r * 32 + n]      = (_Float16)p0;
      pw[r * 32 + 16 + n] = (_Float16)p1;
    }
    // reload P in A-fragment layout (per-wave LDS region, in-order DS)
    V16 pa;
    pa.h[0] = *(const v8h*)(pw + mm * 32 + hi * 8);
    pa.h[1] = *(const v8h*)(pw + mm * 32 + 16 + hi * 8);
    // PV: B[k, e] = vl[e][k] (contiguous along k in LDS)
#pragma unroll
    for (int nt = 0; nt < 12; ++nt) {
      const _Float16* r = vl + (nt * 16 + n) * VPAD + hi * 8;
      V16 vb; vb.h[0] = *(const v8h*)r; vb.h[1] = *(const v8h*)(r + 16);
      acc[nt] = wmma_f16(pa.v, vb.v, acc[nt]);
    }
    __syncthreads();            // safe to overwrite the other buffer
  }

  float inv[8];
#pragma unroll
  for (int j = 0; j < 8; ++j) inv[j] = 1.0f / lrow[j];
#pragma unroll
  for (int nt = 0; nt < 12; ++nt) {
#pragma unroll
    for (int j = 0; j < 8; ++j) {
      int row = b * SS + qrow0 + j + 8 * hi;
      int col = nt * 16 + n;
      oh[(size_t)row * DD + col] = (_Float16)(acc[nt][j] * inv[j]);
    }
  }
}

// ---------------------------------------------------------------------------
// Output projections + gated combine + channel fixups, fp32 out.
// ---------------------------------------------------------------------------
__global__ __launch_bounds__(256)
void combine_kernel(const float* __restrict__ x, const _Float16* __restrict__ ohr,
                    const _Float16* __restrict__ ohw, const _Float16* __restrict__ wh,
                    const float* __restrict__ fl, float* __restrict__ out) {
  const _Float16* WoR = wh + (size_t)3 * DD * DD;
  const _Float16* WoW = wh + (size_t)7 * DD * DD;
  int lane = threadIdx.x & 31, wave = threadIdx.x >> 5;
  int row0 = (blockIdx.x * 8 + wave) * 16;
  int b = row0 >> 12;
  float rg = fl[b], wg = fl[NB + b], ready = fl[2 * NB + b];
  float xg = 1.0f - rg - wg;
  int n = lane & 15, hi = lane >> 4;

  v16h ar[6], aw[6];
#pragma unroll
  for (int ks = 0; ks < 6; ++ks) {
    ar[ks] = loadA(ohr, DD, row0, ks * 32, lane);
    aw[ks] = loadA(ohw, DD, row0, ks * 32, lane);
  }
#pragma unroll
  for (int nt = 0; nt < 12; ++nt) {
    v8f pr = {}, pv = {};
#pragma unroll
    for (int ks = 0; ks < 6; ++ks)
      pr = wmma_f16(ar[ks], loadB_NT(WoR, DD, nt * 16, ks * 32, lane), pr);
#pragma unroll
    for (int ks = 0; ks < 6; ++ks)
      pv = wmma_f16(aw[ks], loadB_NT(WoW, DD, nt * 16, ks * 32, lane), pv);
#pragma unroll
    for (int j = 0; j < 8; ++j) {
      int row = row0 + j + 8 * hi;
      int col = nt * 16 + n;
      float val = xg * x[(size_t)row * DD + col] + rg * pr[j] + wg * pv[j];
      if (col == CH_READ || col == CH_WRITE) val = 0.0f;
      if (col == CH_READY) val = ready;
      out[(size_t)row * DD + col] = val;
    }
  }
}

// ---------------------------------------------------------------------------
// Host launcher
// ---------------------------------------------------------------------------
extern "C" void kernel_launch(void* const* d_in, const int* in_sizes, int n_in,
                              void* d_out, int out_size, void* d_ws, size_t ws_size,
                              hipStream_t stream) {
  const float* x = (const float*)d_in[0];
  const float* W[8];
  for (int i = 0; i < 8; ++i) W[i] = (const float*)d_in[1 + i];
  float* out = (float*)d_out;

  char* ws = (char*)d_ws;
  size_t off = 0;
  auto alloc = [&](size_t bytes) -> char* {
    char* p = ws + off;
    off += (bytes + 255) & ~(size_t)255;
    return p;
  };
  const size_t XB = (size_t)NROWS * DD * sizeof(_Float16);
  _Float16* xh   = (_Float16*)alloc(XB);
  _Float16* wh   = (_Float16*)alloc((size_t)8 * DD * DD * sizeof(_Float16));
  _Float16* q_r  = (_Float16*)alloc(XB);
  _Float16* k_r  = (_Float16*)alloc(XB);
  _Float16* vT_r = (_Float16*)alloc(XB);
  _Float16* q_w  = (_Float16*)alloc(XB);
  _Float16* k_w  = (_Float16*)alloc(XB);
  _Float16* vT_w = (_Float16*)alloc(XB);
  _Float16* o_r  = (_Float16*)alloc(XB);
  _Float16* o_w  = (_Float16*)alloc(XB);
  float* fl      = (float*)alloc(3 * NB * sizeof(float));

  // 1) convert x and weights to f16
  {
    int n = NROWS * DD;
    cvt_f32_f16<<<(n / 4 + 255) / 256, 256, 0, stream>>>(x, xh, n);
    int wn = DD * DD;
    for (int i = 0; i < 8; ++i)
      cvt_f32_f16<<<(wn / 4 + 255) / 256, 256, 0, stream>>>(W[i], wh + (size_t)i * wn, wn);
  }
  // 2) gate flags
  flags_kernel<<<1, 32, 0, stream>>>(x, fl);
  // 3) QKV projections (6 GEMMs)
  qkv_kernel<<<dim3(NROWS / 128, 6), 256, 0, stream>>>(xh, wh, q_r, k_r, vT_r,
                                                       q_w, k_w, vT_w);
  // 4) dual flash attention (async LDS staging)
  attn_kernel<<<dim3(SS / 128, NB, 2), 256, 0, stream>>>(q_r, k_r, vT_r, o_r,
                                                         q_w, k_w, vT_w, o_w);
  // 5) output projections + gated combine + channel fixups
  combine_kernel<<<NROWS / 128, 256, 0, stream>>>(x, o_r, o_w, wh, fl, out);
}
